// NATVisionEncoder_68925635166373
// MI455X (gfx1250) — matrix operand order repaired and checked by reading, not compiled, not run on hardware
//
#include <hip/hip_runtime.h>
#include <hip/hip_bf16.h>

// ---------------- problem constants ----------------
#define B_      32
#define IMG_    224
#define P_      16
#define DIM_    768
#define DEPTH_  12
#define HEADS_  12
#define SEQ_    196     // (224/16)^2
#define MAXSEQ_ 128
#define DFF_    3072
#define HD_     64
#define MAXREP_ 16
#define NTOK_   (B_ * SEQ_)   // 6272
#define SP_     224           // SEQ padded to multiple of 32 for WMMA K

typedef __bf16 bf16;
typedef __attribute__((ext_vector_type(16))) __bf16 v16bf;
typedef __attribute__((ext_vector_type(8)))  __bf16 v8bf;
typedef __attribute__((ext_vector_type(8)))  float  v8f;
typedef __attribute__((ext_vector_type(4)))  int    v4i;

// ---- CDNA5 async global->LDS copy (ASYNCcnt) with safe fallback ----
#if defined(__has_builtin)
#if __has_builtin(__builtin_amdgcn_global_load_async_to_lds_b128)
#define ASYNC_LDS 1
#endif
#endif

__device__ __forceinline__ void cp_lds16(void* lds, const void* g) {
#ifdef ASYNC_LDS
  v4i* gg = (v4i*)g;    // strip const, still generic
  v4i* ll = (v4i*)lds;
  __builtin_amdgcn_global_load_async_to_lds_b128(
      (__attribute__((address_space(1))) v4i*)gg,
      (__attribute__((address_space(3))) v4i*)ll, 0, 0);
#else
  *(v8bf*)lds = *(const v8bf*)g;
#endif
}
__device__ __forceinline__ void async_wait4() {
#ifdef ASYNC_LDS
#if __has_builtin(__builtin_amdgcn_s_wait_asynccnt)
  __builtin_amdgcn_s_wait_asynccnt(4);
#else
  asm volatile("s_wait_asynccnt 0x4" ::: "memory");
#endif
#endif
}
__device__ __forceinline__ void async_wait0() {
#ifdef ASYNC_LDS
#if __has_builtin(__builtin_amdgcn_s_wait_asynccnt)
  __builtin_amdgcn_s_wait_asynccnt(0);
#else
  asm volatile("s_wait_asynccnt 0x0" ::: "memory");
#endif
#endif
}

union FragU { v16bf v; v8bf h[2]; };

// A-fragment (16-bit, 16x32) per ISA 7.12.2: lane row M=(lane&15),
// kbase=8*(lane>=16); two 16B LDS reads at kbase and kbase+16.
__device__ __forceinline__ v16bf frag_a16(const bf16* row_k0, int kbase) {
  FragU u;
  u.h[0] = *(const v8bf*)(row_k0 + kbase);
  u.h[1] = *(const v8bf*)(row_k0 + kbase + 16);
  return u.v;
}
// B-fragment (16-bit, 32x16): lane column N=(lane&15), K at kbase..+7 and kbase+16..+23 (stride ld).
__device__ __forceinline__ v16bf frag_b16(const bf16* p, int ld, int kbase) {
  FragU u;
#pragma unroll
  for (int i = 0; i < 8; ++i) {
    u.v[i] = p[(kbase + i) * ld];
    u.v[8 + i] = p[(kbase + 16 + i) * ld];
  }
  return u.v;
}
__device__ __forceinline__ v8f wmma_bf16(v16bf a, v16bf b, v8f c) {
  return __builtin_amdgcn_wmma_f32_16x16x32_bf16(false, a, false, b,
                                                 (short)0, c, false, false);
}

// ---------------- f32 -> bf16 conversion (weights, once per launch) --------
__global__ __launch_bounds__(256) void cvt_bf16_kernel(
    const float* __restrict__ src, bf16* __restrict__ dst, int n) {
  int i = blockIdx.x * 256 + threadIdx.x;
  if (i < n) dst[i] = (bf16)src[i];
}

// ---------------- patchify: x[B,3,224,224] -> Xp_bf[NTOK, 768] -------------
__global__ __launch_bounds__(256) void patchify_kernel(
    const float* __restrict__ x, bf16* __restrict__ Xp) {
  int i = blockIdx.x * 256 + threadIdx.x;
  if (i >= NTOK_ * DIM_) return;
  int row = i / DIM_, col = i % DIM_;
  int b = row / SEQ_, s = row % SEQ_;
  int ph = s / (IMG_ / P_), pw = s % (IMG_ / P_);
  int c = col / (P_ * P_), rem = col % (P_ * P_);
  int pr = rem / P_, pc = rem % P_;
  Xp[i] = (bf16)x[((size_t)(b * 3 + c) * IMG_ + ph * P_ + pr) * IMG_ + pw * P_ + pc];
}

// ---------------- h = patch + pos (f32 + bf16 shadow) ----------------------
__global__ __launch_bounds__(256) void add_pos_kernel(
    const float* __restrict__ patch, const float* __restrict__ pos,
    float* __restrict__ h, bf16* __restrict__ hb) {
  int i = blockIdx.x * 256 + threadIdx.x;
  if (i >= NTOK_ * DIM_) return;
  float v = patch[i] + pos[i % (SEQ_ * DIM_)];
  h[i] = v;
  hb[i] = (bf16)v;
}

// ---------------- WMMA GEMM: C = A[MxK](bf16) * B[KxN](bf16) + bias --------
// Double-buffered LDS tiles filled via async global->LDS b128 copies.
// Optional GELU; writes f32 C and/or bf16 Cb. M,N multiples of 128, K of 32.
__global__ __launch_bounds__(256) void gemm_bias_kernel(
    const bf16* __restrict__ A, const bf16* __restrict__ Bw,
    const float* __restrict__ bias, float* __restrict__ C,
    bf16* __restrict__ Cb, int M, int N, int K, int fuse_gelu) {
  __shared__ __align__(16) bf16 As[2][128][32];
  __shared__ __align__(16) bf16 Bs[2][32][128];

  const int tid  = threadIdx.x;
  const int lane = tid & 31, wave = tid >> 5;
  const int wrow = wave >> 2, wcol = wave & 3;       // 2x4 wave grid
  const int m0 = blockIdx.y * 128, n0 = blockIdx.x * 128;
  const int lo    = lane & 15;
  const int hi8   = (lane >= 16) ? 8 : 0;
  const int kbase = hi8;

  // staging chunk coords (16B = 8 bf16 per chunk; 2 A chunks + 2 B chunks/thread)
  const int a0m = tid >> 2,         a0k = (tid & 3) * 8;
  const int a1m = (tid + 256) >> 2, a1k = (tid & 3) * 8;
  const int b0k = tid >> 4,         b0n = (tid & 15) * 8;
  const int b1k = (tid + 256) >> 4, b1n = (tid & 15) * 8;

  v8f acc[4][2] = {};   // 64 rows x 32 cols per wave

#define STAGE(buf, k0)                                                        \
  do {                                                                        \
    cp_lds16(&As[buf][a0m][a0k], A + (size_t)(m0 + a0m) * K + (k0) + a0k);    \
    cp_lds16(&As[buf][a1m][a1k], A + (size_t)(m0 + a1m) * K + (k0) + a1k);    \
    cp_lds16(&Bs[buf][b0k][b0n], Bw + (size_t)((k0) + b0k) * N + n0 + b0n);   \
    cp_lds16(&Bs[buf][b1k][b1n], Bw + (size_t)((k0) + b1k) * N + n0 + b1n);   \
  } while (0)

  const int nk = K / 32;
  STAGE(0, 0);
  if (nk > 1) STAGE(1, 32);

  for (int t = 0; t < nk; ++t) {
    if (t + 1 < nk) async_wait4(); else async_wait0();
    __syncthreads();
    const int buf = t & 1;
    v16bf af[4];
#pragma unroll
    for (int i = 0; i < 4; ++i)
      af[i] = frag_a16(&As[buf][wrow * 64 + i * 16 + lo][0], kbase);
    v16bf bfr[2];
#pragma unroll
    for (int j = 0; j < 2; ++j)
      bfr[j] = frag_b16(&Bs[buf][0][wcol * 32 + j * 16 + lo], 128, kbase);
#pragma unroll
    for (int i = 0; i < 4; ++i)
#pragma unroll
      for (int j = 0; j < 2; ++j)
        acc[i][j] = wmma_bf16(af[i], bfr[j], acc[i][j]);
    __syncthreads();
    if (t + 2 < nk) STAGE(buf, (t + 2) * 32);
  }
#undef STAGE

#pragma unroll
  for (int i = 0; i < 4; ++i) {
    const int mb = m0 + wrow * 64 + i * 16 + hi8;
#pragma unroll
    for (int j = 0; j < 2; ++j) {
      const int n = n0 + wcol * 32 + j * 16 + lo;
      const float bb = bias ? bias[n] : 0.f;
#pragma unroll
      for (int r = 0; r < 8; ++r) {
        float v = acc[i][j][r] + bb;
        if (fuse_gelu) {
          float t2 = v;
          v = 0.5f * t2 * (1.f + tanhf(0.7978845608f * (t2 + 0.044715f * t2 * t2 * t2)));
        }
        if (C)  C[(size_t)(mb + r) * N + n] = v;
        if (Cb) Cb[(size_t)(mb + r) * N + n] = (bf16)v;
      }
    }
  }
}

// ---------------- attention: softmax(QK^T/8) V, per (qtile, head, batch) ---
__global__ __launch_bounds__(256) void attention_kernel(
    const bf16* __restrict__ qkv, bf16* __restrict__ o) {
  __shared__ __align__(16) bf16  Qs[32][72];    // [qrow][dim]
  __shared__ __align__(16) bf16  KVs[64][72];   // K phase: [dim][seq]; V: [seq][dim]
  __shared__ __align__(16) float Sc[32][SP_];   // scores f32
  __shared__ __align__(16) bf16  Pa[32][SP_];   // probs bf16

  const int qt = blockIdx.x, hh = blockIdx.y, b = blockIdx.z;
  const int tid = threadIdx.x, lane = tid & 31, wave = tid >> 5;
  const int qsub = wave >> 2, wq = wave & 3;          // 2x4 wave grid
  const int lo = lane & 15, hi8 = (lane >= 16) ? 8 : 0, kbase = hi8;
  const int qs0 = qt * 32;
  const float scale = 0.125f;                          // 1/sqrt(64)
  const int qoff = hh * HD_, koff = DIM_ + hh * HD_, voff = 2 * DIM_ + hh * HD_;
  const v8bf zb = {};

  {  // load Q tile (32 x 64), zero-pad rows >= SEQ
    int rr = tid >> 3, dseg = (tid & 7) * 8;
    int s = qs0 + rr;
    v8bf q = zb;
    if (s < SEQ_)
      q = *(const v8bf*)(qkv + (size_t)(b * SEQ_ + s) * (3 * DIM_) + qoff + dseg);
    *(v8bf*)&Qs[rr][dseg] = q;
  }

  // phase 1: scores = scale * Q K^T   (K staged transposed: KVs[dim][seq])
  for (int g = 0; g < 4; ++g) {
    __syncthreads();
    {
      int sq = tid >> 2, dseg = (tid & 3) * 16;
      int s = g * 64 + sq;
      v8bf klo = zb, khi = zb;
      if (s < SEQ_) {
        const bf16* p = qkv + (size_t)(b * SEQ_ + s) * (3 * DIM_) + koff + dseg;
        klo = *(const v8bf*)p;
        khi = *(const v8bf*)(p + 8);
      }
#pragma unroll
      for (int i = 0; i < 8; ++i) KVs[dseg + i][sq] = klo[i];
#pragma unroll
      for (int i = 0; i < 8; ++i) KVs[dseg + 8 + i][sq] = khi[i];
    }
    __syncthreads();
    v8f acc = {};
#pragma unroll
    for (int ks = 0; ks < 2; ++ks) {
      v16bf a  = frag_a16(&Qs[qsub * 16 + lo][ks * 32], kbase);
      v16bf bb = frag_b16(&KVs[ks * 32][wq * 16 + lo], 72, kbase);
      acc = wmma_bf16(a, bb, acc);
    }
#pragma unroll
    for (int r = 0; r < 8; ++r)
      Sc[qsub * 16 + r + hi8][g * 64 + wq * 16 + lo] = acc[r] * scale;
  }
  __syncthreads();

  // phase 2: row softmax (8 lanes per row), write bf16 probs, pad -> 0
  {
    int row = tid >> 3, sub = tid & 7;
    float mx = -1e30f;
    for (int c = sub; c < SEQ_; c += 8) mx = fmaxf(mx, Sc[row][c]);
    for (int d = 1; d < 8; d <<= 1) mx = fmaxf(mx, __shfl_xor(mx, d, 32));
    float sm = 0.f;
    for (int c = sub; c < SEQ_; c += 8) sm += __expf(Sc[row][c] - mx);
    for (int d = 1; d < 8; d <<= 1) sm += __shfl_xor(sm, d, 32);
    float inv = 1.f / sm;
    for (int c = sub; c < SP_; c += 8)
      Pa[row][c] = (c < SEQ_) ? (bf16)(__expf(Sc[row][c] - mx) * inv) : (bf16)0.f;
  }

  // phase 3: O = P V   (V staged natural: KVs[seq][dim])
  v8f oacc = {};
  for (int g = 0; g < 4; ++g) {
    __syncthreads();
    {
      int sq = tid >> 2, dseg = (tid & 3) * 16;
      int s = g * 64 + sq;
      v8bf vlo = zb, vhi = zb;
      if (s < SEQ_) {
        const bf16* p = qkv + (size_t)(b * SEQ_ + s) * (3 * DIM_) + voff + dseg;
        vlo = *(const v8bf*)p;
        vhi = *(const v8bf*)(p + 8);
      }
      *(v8bf*)&KVs[sq][dseg] = vlo;
      *(v8bf*)&KVs[sq][dseg + 8] = vhi;
    }
    __syncthreads();
#pragma unroll
    for (int ks = 0; ks < 2; ++ks) {
      v16bf a  = frag_a16(&Pa[qsub * 16 + lo][g * 64 + ks * 32], kbase);
      v16bf bb = frag_b16(&KVs[ks * 32][wq * 16 + lo], 72, kbase);
      oacc = wmma_bf16(a, bb, oacc);
    }
  }
#pragma unroll
  for (int r = 0; r < 8; ++r) {
    int s = qs0 + qsub * 16 + r + hi8;
    if (s < SEQ_)
      o[(size_t)(b * SEQ_ + s) * DIM_ + hh * HD_ + wq * 16 + lo] = (bf16)oacc[r];
  }
}

// ---------------- h = LayerNorm(h + r)*s + b  (+ bf16 shadow) --------------
__global__ __launch_bounds__(256) void resid_ln_kernel(
    float* __restrict__ h, const float* __restrict__ r,
    const float* __restrict__ s, const float* __restrict__ b,
    bf16* __restrict__ hb) {
  __shared__ float red[256];
  const int row = blockIdx.x, tid = threadIdx.x;
  float x[3];
  float sum = 0.f;
#pragma unroll
  for (int q = 0; q < 3; ++q) {
    int c = tid + q * 256;
    x[q] = h[(size_t)row * DIM_ + c] + r[(size_t)row * DIM_ + c];
    sum += x[q];
  }
  red[tid] = sum; __syncthreads();
  for (int st = 128; st > 0; st >>= 1) {
    if (tid < st) red[tid] += red[tid + st];
    __syncthreads();
  }
  float mean = red[0] * (1.f / DIM_);
  __syncthreads();
  float vs = 0.f;
#pragma unroll
  for (int q = 0; q < 3; ++q) { float d = x[q] - mean; vs += d * d; }
  red[tid] = vs; __syncthreads();
  for (int st = 128; st > 0; st >>= 1) {
    if (tid < st) red[tid] += red[tid + st];
    __syncthreads();
  }
  float inv = rsqrtf(red[0] * (1.f / DIM_) + 1e-5f);
#pragma unroll
  for (int q = 0; q < 3; ++q) {
    int c = tid + q * 256;
    float v = (x[q] - mean) * inv * s[c] + b[c];
    h[(size_t)row * DIM_ + c] = v;
    hb[(size_t)row * DIM_ + c] = (bf16)v;
  }
}

// ---------------- counts = argmax(h @ rep_w + rep_b), 16 rows/block --------
__global__ __launch_bounds__(256) void rep_argmax_kernel(
    const float* __restrict__ h, const float* __restrict__ w,
    const float* __restrict__ bias, int* __restrict__ counts) {
  int row = blockIdx.x * 16 + (threadIdx.x >> 4);
  int j = threadIdx.x & 15;
  const float* hr = h + (size_t)row * DIM_;
  float acc = bias[j];
  for (int k = 0; k < DIM_; ++k) acc += hr[k] * w[k * MAXREP_ + j];
  float bv = acc; int bi = j;
  for (int d = 1; d < 16; d <<= 1) {
    float ov = __shfl_xor(bv, d, 32);
    int   oi = __shfl_xor(bi, d, 32);
    if (ov > bv || (ov == bv && oi < bi)) { bv = ov; bi = oi; }
  }
  if (j == 0) counts[row] = bi;
}

// ---------------- cumsum + searchsorted + gather patch -> query ------------
__global__ __launch_bounds__(256) void pack_query_kernel(
    const int* __restrict__ counts, const float* __restrict__ patch,
    float* __restrict__ outq) {
  __shared__ int cs[SEQ_];
  __shared__ int idxs[MAXSEQ_];
  const int b = blockIdx.x, tid = threadIdx.x;
  if (tid == 0) {
    int acc = 0;
    for (int s = 0; s < SEQ_; ++s) { acc += counts[b * SEQ_ + s]; cs[s] = acc; }
  }
  __syncthreads();
  if (tid < MAXSEQ_) {
    int t = tid;
    if (t < cs[SEQ_ - 1]) {   // first j with cs[j] > t
      int loi = 0, hii = SEQ_ - 1;
      while (loi < hii) { int mid = (loi + hii) >> 1; if (cs[mid] > t) hii = mid; else loi = mid + 1; }
      idxs[t] = loi;
    } else idxs[t] = -1;
  }
  __syncthreads();
  for (int e = tid; e < MAXSEQ_ * DIM_; e += 256) {
    int t = e / DIM_, c = e % DIM_;
    int idx = idxs[t];
    outq[(size_t)b * MAXSEQ_ * DIM_ + e] =
        (idx >= 0) ? patch[((size_t)b * SEQ_ + idx) * DIM_ + c] : 0.f;
  }
}

// ---------------- orchestration ----------------
extern "C" void kernel_launch(void* const* d_in, const int* in_sizes, int n_in,
                              void* d_out, int out_size, void* d_ws, size_t ws_size,
                              hipStream_t stream) {
  const float* x       = (const float*)d_in[0];
  const float* patch_w = (const float*)d_in[1];
  const float* patch_b = (const float*)d_in[2];
  const float* pos     = (const float*)d_in[3];
  const float* Wqkv    = (const float*)d_in[4];
  const float* bqkv    = (const float*)d_in[5];
  const float* Wo      = (const float*)d_in[6];
  const float* bo      = (const float*)d_in[7];
  const float* ln1s    = (const float*)d_in[8];
  const float* ln1b    = (const float*)d_in[9];
  const float* W1      = (const float*)d_in[10];
  const float* b1      = (const float*)d_in[11];
  const float* W2      = (const float*)d_in[12];
  const float* b2      = (const float*)d_in[13];
  const float* ln2s    = (const float*)d_in[14];
  const float* ln2b    = (const float*)d_in[15];
  const float* rep_w   = (const float*)d_in[16];
  const float* rep_b   = (const float*)d_in[17];

  char* wsb = (char*)d_ws;
  size_t off = 0;
  auto alloc = [&](size_t bytes) -> void* {
    void* p = wsb + off;
    off += (bytes + 255) & ~(size_t)255;
    return p;
  };
  // f32 state
  float* patch  = (float*)alloc((size_t)NTOK_ * DIM_ * 4);
  float* h      = (float*)alloc((size_t)NTOK_ * DIM_ * 4);
  float* tmp    = (float*)alloc((size_t)NTOK_ * DIM_ * 4);
  int*   counts = (int*)alloc((size_t)NTOK_ * 4);
  // bf16 activations
  bf16* Xp_bf   = (bf16*)alloc((size_t)NTOK_ * DIM_ * 2);
  bf16* h_bf    = (bf16*)alloc((size_t)NTOK_ * DIM_ * 2);
  bf16* qkv_bf  = (bf16*)alloc((size_t)NTOK_ * 3 * DIM_ * 2);
  bf16* o_bf    = (bf16*)alloc((size_t)NTOK_ * DIM_ * 2);
  bf16* ff_bf   = (bf16*)alloc((size_t)NTOK_ * DFF_ * 2);
  // bf16 weights (converted once per launch; ~171MB total, L2-resident)
  bf16* pw_bf   = (bf16*)alloc((size_t)DIM_ * DIM_ * 2);
  bf16* Wqkv_bf = (bf16*)alloc((size_t)DEPTH_ * DIM_ * 3 * DIM_ * 2);
  bf16* Wo_bf   = (bf16*)alloc((size_t)DEPTH_ * DIM_ * DIM_ * 2);
  bf16* W1_bf   = (bf16*)alloc((size_t)DEPTH_ * DIM_ * DFF_ * 2);
  bf16* W2_bf   = (bf16*)alloc((size_t)DEPTH_ * DFF_ * DIM_ * 2);

  const dim3 blk(256);
  const int elems = NTOK_ * DIM_;
  auto cvt = [&](const float* s, bf16* d, size_t n) {
    cvt_bf16_kernel<<<dim3((unsigned)((n + 255) / 256)), blk, 0, stream>>>(s, d, (int)n);
  };
  cvt(patch_w, pw_bf,   (size_t)DIM_ * DIM_);
  cvt(Wqkv,    Wqkv_bf, (size_t)DEPTH_ * DIM_ * 3 * DIM_);
  cvt(Wo,      Wo_bf,   (size_t)DEPTH_ * DIM_ * DIM_);
  cvt(W1,      W1_bf,   (size_t)DEPTH_ * DIM_ * DFF_);
  cvt(W2,      W2_bf,   (size_t)DEPTH_ * DFF_ * DIM_);

  patchify_kernel<<<dim3((elems + 255) / 256), blk, 0, stream>>>(x, Xp_bf);
  gemm_bias_kernel<<<dim3(DIM_ / 128, NTOK_ / 128), blk, 0, stream>>>(
      Xp_bf, pw_bf, patch_b, patch, nullptr, NTOK_, DIM_, DIM_, 0);
  add_pos_kernel<<<dim3((elems + 255) / 256), blk, 0, stream>>>(patch, pos, h, h_bf);

  for (int l = 0; l < DEPTH_; ++l) {
    gemm_bias_kernel<<<dim3(3 * DIM_ / 128, NTOK_ / 128), blk, 0, stream>>>(
        h_bf, Wqkv_bf + (size_t)l * DIM_ * 3 * DIM_, bqkv + (size_t)l * 3 * DIM_,
        nullptr, qkv_bf, NTOK_, 3 * DIM_, DIM_, 0);
    attention_kernel<<<dim3((SEQ_ + 31) / 32, HEADS_, B_), blk, 0, stream>>>(qkv_bf, o_bf);
    gemm_bias_kernel<<<dim3(DIM_ / 128, NTOK_ / 128), blk, 0, stream>>>(
        o_bf, Wo_bf + (size_t)l * DIM_ * DIM_, bo + (size_t)l * DIM_,
        tmp, nullptr, NTOK_, DIM_, DIM_, 0);
    resid_ln_kernel<<<dim3(NTOK_), blk, 0, stream>>>(
        h, tmp, ln1s + (size_t)l * DIM_, ln1b + (size_t)l * DIM_, h_bf);
    gemm_bias_kernel<<<dim3(DFF_ / 128, NTOK_ / 128), blk, 0, stream>>>(
        h_bf, W1_bf + (size_t)l * DIM_ * DFF_, b1 + (size_t)l * DFF_,
        nullptr, ff_bf, NTOK_, DFF_, DIM_, 1);
    gemm_bias_kernel<<<dim3(DIM_ / 128, NTOK_ / 128), blk, 0, stream>>>(
        ff_bf, W2_bf + (size_t)l * DFF_ * DIM_, b2 + (size_t)l * DIM_,
        tmp, nullptr, NTOK_, DIM_, DFF_, 0);
    resid_ln_kernel<<<dim3(NTOK_), blk, 0, stream>>>(
        h, tmp, ln2s + (size_t)l * DIM_, ln2b + (size_t)l * DIM_, h_bf);
  }

  // out_state
  (void)hipMemcpyAsync(d_out, h, (size_t)NTOK_ * DIM_ * 4,
                       hipMemcpyDeviceToDevice, stream);
  // repeat head + packed query
  rep_argmax_kernel<<<dim3(NTOK_ / 16), blk, 0, stream>>>(h, rep_w, rep_b, counts);
  float* outq = (float*)d_out + (size_t)NTOK_ * DIM_;
  pack_query_kernel<<<dim3(B_), blk, 0, stream>>>(counts, patch, outq);
}
